// PagedAttention_57672820850768
// MI455X (gfx1250) — matrix operand order repaired and checked by reading, not compiled, not run on hardware
//
#include <hip/hip_runtime.h>

// ---------------------------------------------------------------------------
// MHA forward for MI455X (gfx1250, wave32, WMMA).
// HIDDEN=2048, N_HEADS=16, HEAD_DIM=128, BATCH=2, SEQ=2048.
// Pipeline: f32->bf16 convert, QKV GEMMs (WMMA bf16, async-LDS double
// buffered), causal flash-attention (WMMA bf16 + f32 online softmax),
// O GEMM (WMMA bf16 -> f32 out).
// ---------------------------------------------------------------------------

#define HIDDEN   2048
#define N_HEADS  16
#define HEAD_DIM 128
#define BATCH    2
#define SEQ      2048
#define MROWS    (BATCH * SEQ)   // 4096

typedef __bf16 bf16;
typedef __attribute__((ext_vector_type(16))) __bf16 v16bf;
typedef __attribute__((ext_vector_type(8)))  __bf16 v8bf;
typedef __attribute__((ext_vector_type(8)))  float  v8f;

// --- WMMA fragment helpers (contiguous k order used by this code, not the ISA table order; wave32) ----------
// A-matrix 16x32 bf16: lane l holds row m=l&15; half h=l>>4 holds
// K = h*16 + (0..15) contiguously (same order as B fragments).
__device__ __forceinline__ v16bf load_a_frag(const bf16* rowp, int hf) {
  const v8bf lo = *reinterpret_cast<const v8bf*>(rowp + hf * 16);
  const v8bf hi = *reinterpret_cast<const v8bf*>(rowp + hf * 16 + 8);
  v16bf a;
#pragma unroll
  for (int i = 0; i < 8; ++i) { a[i] = lo[i]; a[i + 8] = hi[i]; }
  return a;
}

// B-matrix 32x16 bf16: lane l holds col n=l&15; half h=l>>4 holds
// K = h*16 + (0..15) contiguously.
__device__ __forceinline__ v16bf load_b_frag(const bf16* p) {
  return *reinterpret_cast<const v16bf*>(p);
}

__device__ __forceinline__ v8f wmma_bf16(v16bf a, v16bf b, v8f c) {
  // (neg_a, A, neg_b, B, c_mod, C, reuse_a, reuse_b)
  return __builtin_amdgcn_wmma_f32_16x16x32_bf16(false, a, false, b,
                                                 (short)0, c, false, false);
}

// --- CDNA5 async global->LDS DMA (GLOBAL_LOAD_ASYNC_TO_LDS_B128, ASYNCcnt) -
// VDST VGPR carries the LDS byte address; flat->LDS mapping uses addr[31:0].
__device__ __forceinline__ void async_load_b128(bf16* ldsDst, const bf16* gSrc) {
  const unsigned ldsAddr = (unsigned)(size_t)(void*)ldsDst;
  asm volatile("global_load_async_to_lds_b128 %0, %1, off"
               :
               : "v"(ldsAddr), "v"(gSrc)
               : "memory");
}

__device__ __forceinline__ void wait_asynccnt0() {
  asm volatile("s_wait_asynccnt 0x0" ::: "memory");
}

// --- f32 -> bf16 conversion ------------------------------------------------
__global__ void cvt_f32_bf16_kernel(const float* __restrict__ in,
                                    bf16* __restrict__ out, int n) {
  int i = blockIdx.x * blockDim.x + threadIdx.x;
  const int stride = gridDim.x * blockDim.x;
  for (; i < n; i += stride) out[i] = (bf16)in[i];
}

// --- GEMM: Y = X @ W^T + bias (X:[M,K] bf16 rm, W:[N,K] bf16 rm) -----------
// 256 threads = 8 waves; wave tile 32(M) x 64(N); WG tile 128 x 128.
// Double-buffered async global->LDS staging of 128x32 A and B tiles.
// mode 0: f32 out [M,N]      (final projection)
// mode 1: bf16 out [b,h,s,d] (Q with scale folded in, K)
// mode 2: bf16 out [b,h,d,s] (V transposed for P*V B-fragments)
__global__ __launch_bounds__(256) void gemm_bf16_kernel(
    const bf16* __restrict__ A, const bf16* __restrict__ W,
    const float* __restrict__ bias, void* __restrict__ out,
    int M, int N, int K, int mode, float scale) {
  // Row pad to 40 elems (80 B) -> b128 fragment reads hit distinct banks.
  __shared__ __align__(16) bf16 As[2][128][40];
  __shared__ __align__(16) bf16 Bs[2][128][40];

  const int tid  = threadIdx.x;
  const int wv   = tid >> 5;
  const int lane = tid & 31;
  const int hf   = lane >> 4;
  const int ln   = lane & 15;
  const int Mblk = blockIdx.y * 128;
  const int Nblk = blockIdx.x * 128;
  const int wm   = (wv >> 1) * 32;   // wave M offset within WG tile
  const int wn   = (wv & 1) * 64;    // wave N offset within WG tile

  // Staging roles: thread t fills 32 B of one row (row = t/2, cols (t&1)*16..)
  const int srow = tid >> 1;
  const int scol = (tid & 1) * 16;
  const bf16* gA = A + (size_t)(Mblk + srow) * K + scol;
  const bf16* gB = W + (size_t)(Nblk + srow) * K + scol;

  v8f acc[2][4] = {};

  // prologue: stage k0 = 0 into buffer 0
  async_load_b128(&As[0][srow][scol], gA);
  async_load_b128(&As[0][srow][scol + 8], gA + 8);
  async_load_b128(&Bs[0][srow][scol], gB);
  async_load_b128(&Bs[0][srow][scol + 8], gB + 8);
  wait_asynccnt0();
  __syncthreads();

  const int nsteps = K / 32;
  for (int step = 0; step < nsteps; ++step) {
    const int buf = step & 1;
    if (step + 1 < nsteps) {  // prefetch next tile while computing this one
      const int k1 = (step + 1) * 32;
      async_load_b128(&As[buf ^ 1][srow][scol], gA + k1);
      async_load_b128(&As[buf ^ 1][srow][scol + 8], gA + k1 + 8);
      async_load_b128(&Bs[buf ^ 1][srow][scol], gB + k1);
      async_load_b128(&Bs[buf ^ 1][srow][scol + 8], gB + k1 + 8);
    }

    v16bf af[2];
#pragma unroll
    for (int mt = 0; mt < 2; ++mt)
      af[mt] = load_a_frag(&As[buf][wm + mt * 16 + ln][0], hf);
    v16bf bfr[4];
#pragma unroll
    for (int nt = 0; nt < 4; ++nt)
      bfr[nt] = load_b_frag(&Bs[buf][wn + nt * 16 + ln][hf * 16]);
#pragma unroll
    for (int mt = 0; mt < 2; ++mt)
#pragma unroll
      for (int nt = 0; nt < 4; ++nt)
        acc[mt][nt] = wmma_bf16(af[mt], bfr[nt], acc[mt][nt]);

    wait_asynccnt0();   // next tile landed
    __syncthreads();    // and everyone is done reading this one
  }

  // Epilogue. C/D layout: lanes 0-15 hold rows r (VGPR r), cols = lane;
  // lanes 16-31 hold rows r+8, cols = lane-16.
#pragma unroll
  for (int mt = 0; mt < 2; ++mt)
#pragma unroll
    for (int nt = 0; nt < 4; ++nt) {
      const int col = Nblk + wn + nt * 16 + ln;
      const float bv = bias[col];
#pragma unroll
      for (int r = 0; r < 8; ++r) {
        const int row = Mblk + wm + mt * 16 + r + 8 * hf;
        const float v = (acc[mt][nt][r] + bv) * scale;
        if (mode == 0) {
          reinterpret_cast<float*>(out)[(size_t)row * N + col] = v;
        } else {
          const int bb = row >> 11;          // row / SEQ
          const int s  = row & (SEQ - 1);
          const int h  = col >> 7;           // col / HEAD_DIM
          const int d  = col & (HEAD_DIM - 1);
          size_t idx;
          if (mode == 1)
            idx = (((size_t)(bb * N_HEADS + h)) * SEQ + s) * HEAD_DIM + d;
          else
            idx = (((size_t)(bb * N_HEADS + h)) * HEAD_DIM + d) * SEQ + s;
          reinterpret_cast<bf16*>(out)[idx] = (bf16)v;
        }
      }
    }
}

// --- Causal flash attention ------------------------------------------------
// Q,K: [B, H, S, D] bf16 (scale folded into Q). Vt: [B, H, D, S] bf16.
// ctx out: [B*S, HIDDEN] bf16. Grid (S/128, B*H), 256 threads = 8 waves,
// each wave owns 16 query rows and streams key blocks of 32.
__global__ __launch_bounds__(256) void attn_kernel(
    const bf16* __restrict__ Q, const bf16* __restrict__ Km,
    const bf16* __restrict__ Vt, bf16* __restrict__ ctx) {
  __shared__ __align__(32) bf16 p_lds[8][16][32];  // per-wave P tile

  const int tid  = threadIdx.x;
  const int wv   = tid >> 5;
  const int lane = tid & 31;
  const int hf   = lane >> 4;
  const int ln   = lane & 15;
  const int bh   = blockIdx.y;                  // b*N_HEADS + h
  const int qb0  = blockIdx.x * 128 + wv * 16;  // first q row of this wave

  const bf16* Qb = Q  + (size_t)bh * SEQ * HEAD_DIM;
  const bf16* Kb = Km + (size_t)bh * SEQ * HEAD_DIM;
  const bf16* Vb = Vt + (size_t)bh * HEAD_DIM * SEQ;

  // Resident Q fragments: 4 k-chunks of 32 over D=128.
  v16bf qf[4];
#pragma unroll
  for (int kk = 0; kk < 4; ++kk)
    qf[kk] = load_a_frag(Qb + (size_t)(qb0 + ln) * HEAD_DIM + kk * 32, hf);

  v8f acc[8] = {};                // 8 d-tiles of 16 -> D=128
  float m_i[8], l_i[8];
#pragma unroll
  for (int r = 0; r < 8; ++r) { m_i[r] = -3.0e38f; l_i[r] = 0.0f; }

  const int nkb = (qb0 + 16 + 31) >> 5;  // causal: key blocks covering <= qb0+15
  for (int kb = 0; kb < nkb; ++kb) {
    const int key0 = kb * 32;

    // scores: S = Q * K^T  (2 key tiles x 4 d-chunks)
    v8f sc[2] = {};
#pragma unroll
    for (int t = 0; t < 2; ++t) {
      const int krow = key0 + t * 16 + ln;
#pragma unroll
      for (int kk = 0; kk < 4; ++kk) {
        const v16bf kf =
            load_b_frag(Kb + (size_t)krow * HEAD_DIM + kk * 32 + hf * 16);
        sc[t] = wmma_bf16(qf[kk], kf, sc[t]);
      }
    }

    // causal mask: element (q = qb0 + r + 8*hf, key = key0 + t*16 + ln)
#pragma unroll
    for (int t = 0; t < 2; ++t) {
      const int key = key0 + t * 16 + ln;
#pragma unroll
      for (int r = 0; r < 8; ++r)
        if (key > qb0 + r + 8 * hf) sc[t][r] = -3.0e38f;
    }

    // online softmax: row reductions across the 16 lanes of each half
    float alpha[8];
#pragma unroll
    for (int r = 0; r < 8; ++r) {
      float v = fmaxf(sc[0][r], sc[1][r]);
      v = fmaxf(v, __shfl_xor(v, 1));
      v = fmaxf(v, __shfl_xor(v, 2));
      v = fmaxf(v, __shfl_xor(v, 4));
      v = fmaxf(v, __shfl_xor(v, 8));
      const float mn = fmaxf(m_i[r], v);
      alpha[r] = __expf(m_i[r] - mn);
      m_i[r] = mn;
    }
#pragma unroll
    for (int r = 0; r < 8; ++r) {
      const float mr = m_i[r];
      const float p0 = __expf(sc[0][r] - mr);
      const float p1 = __expf(sc[1][r] - mr);
      sc[0][r] = p0;
      sc[1][r] = p1;
      float s = p0 + p1;
      s += __shfl_xor(s, 1);
      s += __shfl_xor(s, 2);
      s += __shfl_xor(s, 4);
      s += __shfl_xor(s, 8);
      l_i[r] = l_i[r] * alpha[r] + s;
    }
#pragma unroll
    for (int nt = 0; nt < 8; ++nt)
#pragma unroll
      for (int r = 0; r < 8; ++r) acc[nt][r] *= alpha[r];

    // C-layout P -> LDS -> A-fragment layout (per-wave region, no barrier)
#pragma unroll
    for (int t = 0; t < 2; ++t)
#pragma unroll
      for (int r = 0; r < 8; ++r)
        p_lds[wv][r + 8 * hf][t * 16 + ln] = (bf16)sc[t][r];
    const v16bf pf = load_a_frag(&p_lds[wv][ln][0], hf);

    // acc += P * V  (V transposed: B-frag contiguous along keys)
#pragma unroll
    for (int nt = 0; nt < 8; ++nt) {
      const v16bf vf =
          load_b_frag(Vb + (size_t)(nt * 16 + ln) * SEQ + key0 + hf * 16);
      acc[nt] = wmma_bf16(pf, vf, acc[nt]);
    }
  }

  // normalize and store context [B*S, HIDDEN] bf16
  const int b = bh >> 4;
  const int h = bh & (N_HEADS - 1);
#pragma unroll
  for (int nt = 0; nt < 8; ++nt)
#pragma unroll
    for (int r = 0; r < 8; ++r) {
      const int row = qb0 + r + 8 * hf;
      const int d   = nt * 16 + ln;
      ctx[((size_t)(b * SEQ + row)) * HIDDEN + h * HEAD_DIM + d] =
          (bf16)(acc[nt][r] / l_i[r]);
    }
}

// ---------------------------------------------------------------------------
extern "C" void kernel_launch(void* const* d_in, const int* in_sizes, int n_in,
                              void* d_out, int out_size, void* d_ws,
                              size_t ws_size, hipStream_t stream) {
  (void)in_sizes; (void)n_in; (void)out_size; (void)ws_size;

  const float* hs  = (const float*)d_in[0];
  const float* q_w = (const float*)d_in[1];
  const float* k_w = (const float*)d_in[2];
  const float* v_w = (const float*)d_in[3];
  const float* o_w = (const float*)d_in[4];
  const float* q_b = (const float*)d_in[5];
  const float* k_b = (const float*)d_in[6];
  const float* v_b = (const float*)d_in[7];
  const float* o_b = (const float*)d_in[8];

  // Workspace layout (bf16): x, 4 weights, Q, K, Vt, ctx  (~112 MB total)
  const size_t nX = (size_t)MROWS * HIDDEN;      // 8.39M
  const size_t nW = (size_t)HIDDEN * HIDDEN;     // 4.19M
  bf16* ws   = (bf16*)d_ws;
  bf16* x_bf = ws;                 ws += nX;
  bf16* wq   = ws;                 ws += nW;
  bf16* wk   = ws;                 ws += nW;
  bf16* wvv  = ws;                 ws += nW;
  bf16* wo   = ws;                 ws += nW;
  bf16* Qh   = ws;                 ws += nX;
  bf16* Kh   = ws;                 ws += nX;
  bf16* Vth  = ws;                 ws += nX;
  bf16* ctx  = ws;

  // 1) f32 -> bf16 conversions
  cvt_f32_bf16_kernel<<<2048, 256, 0, stream>>>(hs, x_bf, (int)nX);
  cvt_f32_bf16_kernel<<<2048, 256, 0, stream>>>(q_w, wq, (int)nW);
  cvt_f32_bf16_kernel<<<2048, 256, 0, stream>>>(k_w, wk, (int)nW);
  cvt_f32_bf16_kernel<<<2048, 256, 0, stream>>>(v_w, wvv, (int)nW);
  cvt_f32_bf16_kernel<<<2048, 256, 0, stream>>>(o_w, wo, (int)nW);

  // 2) QKV projections
  const dim3 gg(HIDDEN / 128, MROWS / 128);   // (16, 32)
  const float qscale = 0.08838834764831845f;  // 1/sqrt(HEAD_DIM)
  gemm_bf16_kernel<<<gg, 256, 0, stream>>>(x_bf, wq, q_b, Qh,
                                           MROWS, HIDDEN, HIDDEN, 1, qscale);
  gemm_bf16_kernel<<<gg, 256, 0, stream>>>(x_bf, wk, k_b, Kh,
                                           MROWS, HIDDEN, HIDDEN, 1, 1.0f);
  gemm_bf16_kernel<<<gg, 256, 0, stream>>>(x_bf, wvv, v_b, Vth,
                                           MROWS, HIDDEN, HIDDEN, 2, 1.0f);

  // 3) causal flash attention
  attn_kernel<<<dim3(SEQ / 128, BATCH * N_HEADS), 256, 0, stream>>>(Qh, Kh, Vth,
                                                                    ctx);

  // 4) output projection -> f32 d_out
  gemm_bf16_kernel<<<gg, 256, 0, stream>>>(ctx, wo, o_b, d_out,
                                           MROWS, HIDDEN, HIDDEN, 0, 1.0f);
}
